// GraphConvolution_2095944041230
// MI455X (gfx1250) — matrix operand order repaired and checked by reading, not compiled, not run on hardware
//
#include <hip/hip_runtime.h>

#define IN_F   512
#define OUT_F  512
#define LDS_STRIDE 516   // 512 + 4-float pad: rows land 4 banks apart, multiple of 4 for b128 ops

#define K_CHUNKS   (IN_F / 32)     // 16 K-chunks of 32
#define N_TILES    (OUT_F / 16)    // 32 column tiles of 16

typedef __attribute__((ext_vector_type(16))) __bf16 v16bf;
typedef __attribute__((ext_vector_type(8)))  float  v8f;

// bf16x3 trick: x ~= hi + lo (both bf16); hi*hi + hi*lo + lo*hi ~ fp32 product
__device__ __forceinline__ __bf16 bf_hi(float x) { return (__bf16)x; }
__device__ __forceinline__ __bf16 bf_lo(float x, __bf16 h) { return (__bf16)(x - (float)h); }

// -----------------------------------------------------------------------------
// Kernel 0: pre-pack W into per-lane WMMA B-fragment layout, split hi/lo bf16.
// Fragment f = (chunk * N_TILES + tile); within a fragment: lane 0..31 x 16 bf16.
// B k order used here (matches A): lane = N col (lane&15), K = (lane>>4)*16 + j.
// One thread builds one lane-fragment (16 elements). 16*32*32 = 16384 threads.
// -----------------------------------------------------------------------------
__global__ __launch_bounds__(256)
void pack_w_kernel(const float* __restrict__ W,
                   __bf16* __restrict__ Whi, __bf16* __restrict__ Wlo)
{
    const int f = blockIdx.x * blockDim.x + threadIdx.x;   // 0 .. 16384-1
    const int lane  = f & 31;
    const int tile  = (f >> 5) & (N_TILES - 1);
    const int chunk = f >> 10;

    const int col = tile * 16 + (lane & 15);
    const int k0  = chunk * 32 + (lane >> 4) * 16;

    v16bf hi, lo;
    #pragma unroll
    for (int j = 0; j < 16; ++j) {
        float x = W[(size_t)(k0 + j) * OUT_F + col];
        __bf16 h = bf_hi(x);
        hi[j] = h;
        lo[j] = bf_lo(x, h);
    }
    const size_t off = (size_t)f * 16;
    *(v16bf*)(Whi + off) = hi;
    *(v16bf*)(Wlo + off) = lo;
}

// -----------------------------------------------------------------------------
// Kernel 1: support = X (nrows x 512) * W (512 x 512), fp32 via bf16x3 WMMA.
// Block = 256 threads = 8 waves, one 16-row x 512-col strip per block.
// A tile staged f32 -> LDS (coalesced), then converted ONCE per block into
// packed bf16 A-fragments in LDS; each wave reads a_hi/a_lo via ds_load_b128
// and B fragments via coalesced b128 global loads from the pre-packed Whi/Wlo.
// Hot loop per chunk: 12 x v_wmma + 16 b128 vmem + 4 b128 ds, ~no VALU.
// -----------------------------------------------------------------------------
__global__ __launch_bounds__(256)
void gemm_bf16x3_kernel(const float* __restrict__ A,
                        const __bf16* __restrict__ Whi, const __bf16* __restrict__ Wlo,
                        float* __restrict__ S, int nrows)
{
    __shared__ float  As[16 * LDS_STRIDE];              // 33,024 B
    __shared__ __bf16 AfHi[K_CHUNKS * 32 * 16];         // 16,384 B
    __shared__ __bf16 AfLo[K_CHUNKS * 32 * 16];         // 16,384 B

    const int tid  = threadIdx.x;
    const int lane = tid & 31;
    const int wave = tid >> 5;              // 0..7
    const int m0   = blockIdx.x * 16;       // row tile base

    // ---- stage 1: coalesced f32 copy of the 16x512 A strip into padded LDS ----
    #pragma unroll
    for (int i = 0; i < 8; ++i) {
        int idx = tid + i * 256;            // 0..2047 float4 slots (16 rows * 128)
        int r   = idx >> 7;                 // row 0..15
        int c4  = idx & 127;                // float4 column
        int gr  = m0 + r;
        if (gr >= nrows) gr = nrows - 1;    // clamp (harmless duplicate read)
        float4 v = ((const float4*)A)[(size_t)gr * (IN_F / 4) + c4];
        *(float4*)&As[r * LDS_STRIDE + c4 * 4] = v;
    }
    __syncthreads();

    // ---- stage 2: build bf16 hi/lo A-fragments once per block ----
    // A k order used here (matches B): lane = M row (lane&15),
    //   K = (lane>>4)*16 + j (j=0..15).
    #pragma unroll
    for (int i = 0; i < 2; ++i) {
        int fidx  = tid + i * 256;          // 0..511 = chunk*32 + lane
        int flane = fidx & 31;
        int chunk = fidx >> 5;
        int row   = flane & 15;
        int akb   = (flane >> 4) * 16;
        const float* Ar = &As[row * LDS_STRIDE + chunk * 32];

        v16bf hi, lo;
        #pragma unroll
        for (int j = 0; j < 8; ++j) {
            float x0 = Ar[akb + j];
            __bf16 h0 = bf_hi(x0);
            hi[j] = h0;
            lo[j] = bf_lo(x0, h0);
            float x1 = Ar[8 + akb + j];
            __bf16 h1 = bf_hi(x1);
            hi[j + 8] = h1;
            lo[j + 8] = bf_lo(x1, h1);
        }
        *(v16bf*)&AfHi[fidx * 16] = hi;
        *(v16bf*)&AfLo[fidx * 16] = lo;
    }
    __syncthreads();

    // ---- main loop: 16 chunks x (4 tiles x 3 WMMAs) ----
    const int tile0 = wave * 4;             // wave owns tiles tile0..tile0+3
    v8f acc[4] = {v8f{0.f}, v8f{0.f}, v8f{0.f}, v8f{0.f}};

    for (int c = 0; c < K_CHUNKS; ++c) {
        const v16bf a_hi = *(const v16bf*)&AfHi[(c * 32 + lane) * 16];
        const v16bf a_lo = *(const v16bf*)&AfLo[(c * 32 + lane) * 16];

        #pragma unroll
        for (int t = 0; t < 4; ++t) {
            const size_t foff = (((size_t)c * N_TILES + (tile0 + t)) * 32 + lane) * 16;
            const v16bf b_hi = *(const v16bf*)(Whi + foff);
            const v16bf b_lo = *(const v16bf*)(Wlo + foff);

            // bf16x3: hi*hi + hi*lo + lo*hi  (lo*lo is below fp32 noise)
            acc[t] = __builtin_amdgcn_wmma_f32_16x16x32_bf16(
                         false, a_hi, false, b_hi, (short)0, acc[t], false, false);
            acc[t] = __builtin_amdgcn_wmma_f32_16x16x32_bf16(
                         false, a_hi, false, b_lo, (short)0, acc[t], false, false);
            acc[t] = __builtin_amdgcn_wmma_f32_16x16x32_bf16(
                         false, a_lo, false, b_hi, (short)0, acc[t], false, false);
        }
    }

    // ---- store D tiles: VGPR r -> M = r + 8*(lane>>4), N = lane&15 ----
    const int mro  = (lane >> 4) * 8;
    const int colq = lane & 15;
    #pragma unroll
    for (int t = 0; t < 4; ++t) {
        const int col = (tile0 + t) * 16 + colq;
        #pragma unroll
        for (int r = 0; r < 8; ++r) {
            const int grow = m0 + mro + r;
            if (grow < nrows)
                S[(size_t)grow * OUT_F + col] = acc[t][r];
        }
    }
}

// -----------------------------------------------------------------------------
// Kernel 2: out[n, f] = bias[f]  (float4; also clears harness poison)
// -----------------------------------------------------------------------------
__global__ void init_bias_kernel(float* __restrict__ out, const float* __restrict__ bias,
                                 int total_f4)
{
    int i = blockIdx.x * blockDim.x + threadIdx.x;
    if (i < total_f4) {
        int col = (i << 2) & (OUT_F - 1);
        ((float4*)out)[i] = *(const float4*)(bias + col);
    }
}

// -----------------------------------------------------------------------------
// Kernel 3: for each edge e: out[row[e], :] += support[col[e], :] * val[e]
// 128 threads per edge, float4 gather + 4x global_atomic_add_f32 scatter.
// -----------------------------------------------------------------------------
__global__ __launch_bounds__(256)
void scatter_edges_kernel(const float* __restrict__ S, const float* __restrict__ ev,
                          const int* __restrict__ er, const int* __restrict__ ec,
                          float* __restrict__ out, int nedges)
{
    const int e = blockIdx.x * 2 + (threadIdx.x >> 7);
    if (e >= nedges) return;
    const int t   = threadIdx.x & 127;
    const int src = ec[e];
    const int dst = er[e];
    const float v = ev[e];

    const float4 s = *(const float4*)(S + (size_t)src * OUT_F + t * 4);
    float* o = out + (size_t)dst * OUT_F + t * 4;
    atomicAdd(o + 0, s.x * v);
    atomicAdd(o + 1, s.y * v);
    atomicAdd(o + 2, s.z * v);
    atomicAdd(o + 3, s.w * v);
}

extern "C" void kernel_launch(void* const* d_in, const int* in_sizes, int n_in,
                              void* d_out, int out_size, void* d_ws, size_t ws_size,
                              hipStream_t stream)
{
    const float* input    = (const float*)d_in[0];
    const float* weight   = (const float*)d_in[1];
    const float* bias     = (const float*)d_in[2];
    const float* edge_val = (const float*)d_in[3];
    const int*   edge_row = (const int*)d_in[4];
    const int*   edge_col = (const int*)d_in[5];
    float* out = (float*)d_out;

    const int n_nodes = in_sizes[0] / IN_F;
    const int n_edges = in_sizes[3];

    // Workspace carve-up: [support f32][Whi bf16][Wlo bf16]
    float*  support = (float*)d_ws;                                   // n_nodes*512 f32
    __bf16* Whi     = (__bf16*)((char*)d_ws + (size_t)n_nodes * OUT_F * sizeof(float));
    __bf16* Wlo     = Whi + (size_t)IN_F * OUT_F;                     // 512KB each

    // 0) pre-pack W into WMMA B-fragment layout (hi/lo bf16): 16384 fragments
    pack_w_kernel<<<(K_CHUNKS * N_TILES * 32) / 256, 256, 0, stream>>>(weight, Whi, Wlo);

    // 1) support = X * W  (bf16x3 WMMA)
    const int row_tiles = (n_nodes + 15) / 16;
    gemm_bf16x3_kernel<<<row_tiles, 256, 0, stream>>>(input, Whi, Wlo, support, n_nodes);

    // 2) out = bias (broadcast)
    const int total_f4 = (n_nodes * OUT_F) / 4;
    init_bias_kernel<<<(total_f4 + 255) / 256, 256, 0, stream>>>(out, bias, total_f4);

    // 3) scatter-add edges
    scatter_edges_kernel<<<(n_edges + 1) / 2, 256, 0, stream>>>(
        support, edge_val, edge_row, edge_col, out, n_edges);
}
